// SimpleLSTM_90108413870876
// MI455X (gfx1250) — compile-verified, matrix operand order checked by reading, not run on hardware
//
#include <hip/hip_runtime.h>

typedef __attribute__((ext_vector_type(16))) __bf16 v16bf;
typedef __attribute__((ext_vector_type(8)))  float  v8f;

union Frag32B { v16bf v; uint4 q[2]; };

static __device__ __forceinline__ unsigned short f2bf(float f) {
    unsigned u = __float_as_uint(f);
    u += 0x7FFFu + ((u >> 16) & 1u);        // round-to-nearest-even
    return (unsigned short)(u >> 16);
}
static __device__ __forceinline__ float bf2f(unsigned short h) {
    return __uint_as_float(((unsigned)h) << 16);
}
static __device__ __forceinline__ float sigm(float x) {
    // saturates correctly at +-inf without clamps
    return 1.0f / (1.0f + __expf(-x));
}
static __device__ __forceinline__ float tanh_fast(float x) {
#if __has_builtin(__builtin_amdgcn_tanh_f32)
    return __builtin_amdgcn_tanh_f32(x);     // gfx1250 v_tanh_f32
#elif __has_builtin(__builtin_amdgcn_tanhf)
    return __builtin_amdgcn_tanhf(x);
#else
    x = fminf(fmaxf(x, -15.f), 15.f);
    float e = __expf(2.0f * x);
    return (e - 1.0f) / (e + 1.0f);
#endif
}

// Launder a byte offset (always 0) through a volatile asm so weight loads are
// not loop-invariant across the 420-step time loop (prevents LICM + spill),
// while the *pointer* keeps its global address-space provenance.
static __device__ __forceinline__ long opql(long v) {
    asm volatile("" : "+s"(v));
    return v;
}

constexpr int BT = 32;    // batch rows per workgroup
constexpr int KC = 256;   // concat K = 2H
constexpr int NG = 512;   // 4H gate columns

// One LSTM layer cell for this WG's 32-row batch tile.
// gates(32x512) = [x|h](32x256) @ Wc^T, Wc row-major [512][256] bf16.
// Wave wv owns hidden units [16*wv, 16*wv+16) across all 4 gates.
template<bool DUAL>
static __device__ __forceinline__ void lstm_cell(
    const unsigned short* __restrict__ Wp,   // [512][256] bf16 bits (L2-resident)
    const float* __restrict__ bias,          // [512] combined bias
    const unsigned short* __restrict__ xh,   // LDS [32][256] bf16 bits
    v8f* cst,                                // [2] persistent cell state (C-layout)
    unsigned short* d0,                      // LDS dest (base + column) for h_new
    unsigned short* d1,                      // optional second dest
    int lane, int wv)
{
    const unsigned short* W = Wp + opql(0);  // opaque offset: no cross-step LICM

    v8f acc[2][4] = {};
    const int kh = (lane >= 16) ? 8 : 0;     // K-half per A/B 16-bit layout
    const int m0 = lane & 15;

    #pragma unroll 2
    for (int k0 = 0; k0 < KC; k0 += 32) {
        Frag32B a[2];
        #pragma unroll
        for (int mt = 0; mt < 2; ++mt) {
            const unsigned short* p = xh + (mt * 16 + m0) * KC + k0 + kh;
            a[mt].q[0] = *(const uint4*)p;          // ds_load_b128
            a[mt].q[1] = *(const uint4*)(p + 16);
        }
        Frag32B b[4];
        #pragma unroll
        for (int g = 0; g < 4; ++g) {
            const unsigned short* p = W + (g * 128 + wv * 16 + m0) * KC + k0 + kh;
            b[g].q[0] = *(const uint4*)p;           // global_load_b128 (L2 hit)
            b[g].q[1] = *(const uint4*)(p + 16);
        }
        #pragma unroll
        for (int mt = 0; mt < 2; ++mt)
            #pragma unroll
            for (int g = 0; g < 4; ++g)
                acc[mt][g] = __builtin_amdgcn_wmma_f32_16x16x32_bf16(
                    false, a[mt].v, false, b[g].v, (short)0, acc[mt][g], false, false);
    }

    __syncthreads();   // all waves done reading xh before h_new overwrites it

    const int colj = wv * 16 + m0;
    const float bI = bias[0 * 128 + colj];
    const float bF = bias[1 * 128 + colj];
    const float bG = bias[2 * 128 + colj];
    const float bO = bias[3 * 128 + colj];
    const int rhalf = (lane >= 16) ? 8 : 0;

    #pragma unroll
    for (int mt = 0; mt < 2; ++mt) {
        #pragma unroll
        for (int r = 0; r < 8; ++r) {
            float iv = acc[mt][0][r] + bI;
            float fv = acc[mt][1][r] + bF;
            float gv = acc[mt][2][r] + bG;
            float ov = acc[mt][3][r] + bO;
            float cn = sigm(fv) * cst[mt][r] + sigm(iv) * tanh_fast(gv);
            float hn = sigm(ov) * tanh_fast(cn);
            cst[mt][r] = cn;
            int row = mt * 16 + r + rhalf;          // C-tile layout -> batch row
            unsigned short hb = f2bf(hn);
            d0[row * KC] = hb;
            if (DUAL) d1[row * KC] = hb;
        }
    }
}

__global__ __launch_bounds__(256) __attribute__((amdgpu_waves_per_eu(2)))
void lstm_seq2seq(const float* __restrict__ hist,
                  const float* __restrict__ embW, const float* __restrict__ embb,
                  const float* __restrict__ outW, const float* __restrict__ outb,
                  const unsigned short* __restrict__ Wc,
                  const float* __restrict__ biasc,
                  float* __restrict__ out)
{
    __shared__ __align__(16) unsigned short xh0[BT * KC]; // [x | h0]
    __shared__ __align__(16) unsigned short xh1[BT * KC]; // [h0out | h1]
    __shared__ float red[256];
    __shared__ float pred[BT * 2];

    const int tid  = threadIdx.x;
    const int lane = tid & 31;
    const int wv   = tid >> 5;
    const int wg   = blockIdx.x;
    const int colj = wv * 16 + (lane & 15);

    for (int i = tid; i < BT * 128; i += 256) {      // h,c start at zero
        int row = i >> 7, c = i & 127;
        xh0[row * KC + 128 + c] = 0;
        xh1[row * KC + 128 + c] = 0;
    }
    v8f c0[2] = {}, c1[2] = {};
    __syncthreads();

    const unsigned short* encW0 = Wc + 0 * (NG * KC);
    const unsigned short* encW1 = Wc + 1 * (NG * KC);
    const unsigned short* decW0 = Wc + 2 * (NG * KC);
    const unsigned short* decW1 = Wc + 3 * (NG * KC);
    const float* encB0 = biasc + 0 * NG;
    const float* encB1 = biasc + 1 * NG;
    const float* decB0 = biasc + 2 * NG;
    const float* decB1 = biasc + 3 * NG;

    // ---------------- encoder: 60 steps ----------------
    #pragma unroll 1
    for (int t = 0; t < 60; ++t) {
        {   // embed history[:, t, :] (2 -> 128) into xh0 x-half
            int row = tid >> 3, jb = (tid & 7) * 16;
            int gr = wg * BT + row;
            float p0 = hist[(gr * 60 + t) * 2 + 0];
            float p1 = hist[(gr * 60 + t) * 2 + 1];
            #pragma unroll
            for (int jj = 0; jj < 16; ++jj) {
                int j = jb + jj;
                float v = fmaf(p0, embW[2 * j], fmaf(p1, embW[2 * j + 1], embb[j]));
                xh0[row * KC + j] = f2bf(v);
            }
        }
        __syncthreads();
        lstm_cell<true >(encW0, encB0, xh0, c0, xh0 + 128 + colj, xh1 + colj, lane, wv);
        __syncthreads();
        lstm_cell<false>(encW1, encB1, xh1, c1, xh1 + 128 + colj, nullptr, lane, wv);
        __syncthreads();
    }

    // dec_in0 = zeros
    for (int i = tid; i < BT * 128; i += 256) {
        int row = i >> 7, c = i & 127;
        xh0[row * KC + c] = 0;
    }
    __syncthreads();

    // ---------------- decoder: 360 autoregressive steps ----------------
    #pragma unroll 1
    for (int t = 0; t < 360; ++t) {
        lstm_cell<true >(decW0, decB0, xh0, c0, xh0 + 128 + colj, xh1 + colj, lane, wv);
        __syncthreads();
        lstm_cell<false>(decW1, decB1, xh1, c1, xh1 + 128 + colj, nullptr, lane, wv);
        __syncthreads();

        {   // pred = h_top(32x128) @ outW.T(128x2) + outb : 4 partials per output
            int row = tid >> 3;
            int oc  = (tid >> 2) & 1;
            int part = tid & 3;
            const unsigned short* hrow = xh1 + row * KC + 128 + part * 32;
            const float* wrow = outW + oc * 128 + part * 32;
            float s = 0.f;
            #pragma unroll
            for (int j = 0; j < 32; ++j)
                s = fmaf(bf2f(hrow[j]), wrow[j], s);
            red[tid] = s;
        }
        __syncthreads();
        if (tid < 64) {
            int row = tid >> 1, oc = tid & 1;
            int base = row * 8 + oc * 4;
            float s = red[base] + red[base + 1] + red[base + 2] + red[base + 3] + outb[oc];
            pred[row * 2 + oc] = s;
            out[(wg * BT + row) * (360 * 2) + t * 2 + oc] = s;
        }
        __syncthreads();
        {   // re-embed pred (2 -> 128) into xh0 x-half for next step
            int row = tid >> 3, jb = (tid & 7) * 16;
            float p0 = pred[row * 2 + 0];
            float p1 = pred[row * 2 + 1];
            #pragma unroll
            for (int jj = 0; jj < 16; ++jj) {
                int j = jb + jj;
                float v = fmaf(p0, embW[2 * j], fmaf(p1, embW[2 * j + 1], embb[j]));
                xh0[row * KC + j] = f2bf(v);
            }
        }
        __syncthreads();
    }
}

// Pack [Wih | Whh] -> bf16 Wc[phase][layer][512][256]
__global__ __launch_bounds__(256)
void prep_w(const float* __restrict__ encWih, const float* __restrict__ encWhh,
            const float* __restrict__ decWih, const float* __restrict__ decWhh,
            unsigned short* __restrict__ Wc)
{
    int i = blockIdx.x * 256 + threadIdx.x;   // 524288 total
    int k = i & 255;
    int n = (i >> 8) & 511;
    int l = (i >> 17) & 1;
    int p = (i >> 18) & 1;
    const float* src = (p == 0) ? ((k < 128) ? encWih : encWhh)
                                : ((k < 128) ? decWih : decWhh);
    float v = src[l * (512 * 128) + n * 128 + (k & 127)];
    Wc[(p * 2 + l) * (512 * 256) + n * 256 + k] = f2bf(v);
}

__global__ __launch_bounds__(256)
void prep_b(const float* __restrict__ eb0, const float* __restrict__ eb1,
            const float* __restrict__ db0, const float* __restrict__ db1,
            float* __restrict__ biasc)
{
    int i = blockIdx.x * 256 + threadIdx.x;   // 2048 total
    int j = i & 511;
    int l = (i >> 9) & 1;
    int p = (i >> 10) & 1;
    const float* a = p ? db0 : eb0;
    const float* b = p ? db1 : eb1;
    biasc[(p * 2 + l) * 512 + j] = a[l * 512 + j] + b[l * 512 + j];
}

extern "C" void kernel_launch(void* const* d_in, const int* in_sizes, int n_in,
                              void* d_out, int out_size, void* d_ws, size_t ws_size,
                              hipStream_t stream)
{
    const float* hist   = (const float*)d_in[0];
    const float* embW   = (const float*)d_in[1];
    const float* embb   = (const float*)d_in[2];
    const float* encWih = (const float*)d_in[3];
    const float* encWhh = (const float*)d_in[4];
    const float* encbih = (const float*)d_in[5];
    const float* encbhh = (const float*)d_in[6];
    const float* decWih = (const float*)d_in[7];
    const float* decWhh = (const float*)d_in[8];
    const float* decbih = (const float*)d_in[9];
    const float* decbhh = (const float*)d_in[10];
    const float* outW   = (const float*)d_in[11];
    const float* outb   = (const float*)d_in[12];

    unsigned short* Wc = (unsigned short*)d_ws;
    float* biasc = (float*)((char*)d_ws + (size_t)4 * 512 * 256 * sizeof(unsigned short));

    prep_w<<<2048, 256, 0, stream>>>(encWih, encWhh, decWih, decWhh, Wc);
    prep_b<<<8, 256, 0, stream>>>(encbih, encbhh, decbih, decbhh, biasc);
    lstm_seq2seq<<<4096 / BT, 256, 0, stream>>>(hist, embW, embb, outW, outb,
                                                Wc, biasc, (float*)d_out);
}
